// GraphNetworkGENConv_15178414424349
// MI455X (gfx1250) — matrix operand order repaired and checked by reading, not compiled, not run on hardware
//
#include <hip/hip_runtime.h>
#include <hip/hip_bf16.h>

typedef __attribute__((ext_vector_type(16))) _Float16 v16h;
typedef __attribute__((ext_vector_type(8)))  float    v8f;

#define N_NODES 50000
#define N_EDGES 800000
#define HID 64
#define MSG_EPS 1e-7f
#define LN_EPS  1e-5f

// ---------------------------------------------------------------------------
// Encoder: out[m, 0:32] = relu(in1[m,0:10] @ W1 + b1)
//          out[m,32:64] = relu(in2[m,0:49] @ W2 + b2)
// One block (64 threads) per row m.  Shared between node and edge encoders.
// ---------------------------------------------------------------------------
__global__ void enc_kernel(const float* __restrict__ in1,
                           const float* __restrict__ in2,
                           const float* __restrict__ W1, const float* __restrict__ b1,
                           const float* __restrict__ W2, const float* __restrict__ b2,
                           float* __restrict__ out)
{
    const int m = blockIdx.x;
    const int f = threadIdx.x;            // 0..63
    float acc;
    if (f < 32) {
        acc = b1[f];
        const float* r = in1 + (size_t)m * 10;
        #pragma unroll
        for (int i = 0; i < 10; ++i) acc += r[i] * W1[i * 32 + f];
    } else {
        const int j = f - 32;
        acc = b2[j];
        const float* r = in2 + (size_t)m * 49;
        for (int i = 0; i < 49; ++i) acc += r[i] * W2[i * 32 + j];
    }
    out[(size_t)m * HID + f] = fmaxf(acc, 0.0f);
}

// ---------------------------------------------------------------------------
// Clear per-layer reduction buffers (max / numerator / denominator).
// ---------------------------------------------------------------------------
__global__ void clear_kernel(float* __restrict__ a, float* __restrict__ b,
                             float* __restrict__ c, int n)
{
    int i = blockIdx.x * blockDim.x + threadIdx.x;
    if (i < n) { a[i] = 0.0f; b[i] = 0.0f; c[i] = 0.0f; }
}

// ---------------------------------------------------------------------------
// Edge pass 1: segment-max of s = (relu(h[src]+e)+eps)*t by dst, per feature.
// s > 0 always (eps>0, t=1), so int-bitpattern atomicMax == float max and a
// zero-initialized buffer is a valid identity.
// One block (64 threads) per edge; thread = feature.
// ---------------------------------------------------------------------------
__global__ void edge_max_kernel(const float* __restrict__ h,
                                const float* __restrict__ e,
                                const int*   __restrict__ ei,
                                const float* __restrict__ tptr,
                                float* __restrict__ maxb)
{
    const int eid = blockIdx.x;
    const int f   = threadIdx.x;
    const int src = ei[eid];
    const int dst = ei[N_EDGES + eid];
    const float t = tptr[0];
    const float msg = fmaxf(h[(size_t)src * HID + f] + e[(size_t)eid * HID + f], 0.0f) + MSG_EPS;
    const float s   = msg * t;
    atomicMax((int*)(maxb + (size_t)dst * HID + f), __float_as_int(s));
}

// ---------------------------------------------------------------------------
// Edge pass 2: ex = exp(s - max[dst]); num += msg*ex; den += ex.
// (denominator is constant per (dst,f), so alpha division is deferred.)
// ---------------------------------------------------------------------------
__global__ void edge_sum_kernel(const float* __restrict__ h,
                                const float* __restrict__ e,
                                const int*   __restrict__ ei,
                                const float* __restrict__ tptr,
                                const float* __restrict__ maxb,
                                float* __restrict__ num,
                                float* __restrict__ den)
{
    const int eid = blockIdx.x;
    const int f   = threadIdx.x;
    const int src = ei[eid];
    const int dst = ei[N_EDGES + eid];
    const float t = tptr[0];
    const float msg = fmaxf(h[(size_t)src * HID + f] + e[(size_t)eid * HID + f], 0.0f) + MSG_EPS;
    const float s   = msg * t;
    const float ex  = __expf(s - maxb[(size_t)dst * HID + f]);
    atomicAdd(den + (size_t)dst * HID + f, ex);
    atomicAdd(num + (size_t)dst * HID + f, msg * ex);
}

// ---------------------------------------------------------------------------
// Aggregation finalize: mlp_in = num * rcp(den + 1e-16) + h_in.
// Single v_rcp_f32 (result is rounded to f16 downstream anyway).
// ---------------------------------------------------------------------------
__global__ void aggr_kernel(const float* __restrict__ num,
                            const float* __restrict__ den,
                            const float* __restrict__ hin,
                            float* __restrict__ mlpin, int n)
{
    int i = blockIdx.x * blockDim.x + threadIdx.x;
    if (i < n)
        mlpin[i] = num[i] * __builtin_amdgcn_rcpf(den[i] + 1e-16f) + hin[i];
}

// ---------------------------------------------------------------------------
// Fused GENConv MLP, one block = 16 nodes, 256 threads = 8 wave32.
//   z    = mlp_in @ W1 + b1                             [16 x 128]  (8 waves x wmma)
//   zln  = relu(LayerNorm(z) * g + b)                   (fp32 in LDS -> f16)
//   o    = zln @ W2 + b2                                [16 x 64]   (4 waves x wmma)
//   out  = relu(o + r1 + r2)
// Fragment layouts per CDNA5 ISA 7.12.2 (wave32):
//   A: lane m = lane&15; grp = lane>>4; vgpr v holds K = grp*8+2v(+1) for v<4,
//      16+grp*8+2(v-4)(+1) for v>=4.
//   B: lane n = lane&15; vgpr v holds same K as A.
//   C/D: lane n = lane&15; vgpr v holds row M = v + grp*8.
// ---------------------------------------------------------------------------
__global__ __launch_bounds__(256)
void conv_mlp_kernel(const float* __restrict__ mlpin,
                     const float* __restrict__ W1, const float* __restrict__ b1,
                     const float* __restrict__ lng, const float* __restrict__ lnb,
                     const float* __restrict__ W2, const float* __restrict__ b2,
                     const float* __restrict__ r1, const float* __restrict__ r2,
                     float* __restrict__ out)
{
    __shared__ float    zf[16][128];
    __shared__ _Float16 zh[16][128];

    const int tid  = threadIdx.x;
    const int wave = tid >> 5;
    const int lane = tid & 31;
    const int grp  = lane >> 4;
    const int mn   = lane & 15;                 // A row / B,C col index
    const int nodeA = blockIdx.x * 16 + mn;     // node whose row this lane's A holds

    // Weights are reused by every block: pull them toward the caches.
    __builtin_prefetch(W1, 0, 3);
    __builtin_prefetch(W2, 0, 3);

    // ---------------- GEMM1: mlp_in[16x64] @ W1[64x128] ----------------
    {
        const int c0 = wave * 16;
        v8f acc = {};
        #pragma unroll
        for (int kc = 0; kc < 2; ++kc) {
            v16h a, b;
            #pragma unroll
            for (int v = 0; v < 8; ++v) {
                const int k = ((v < 4) ? (grp * 8 + 2 * v)
                                       : (16 + grp * 8 + 2 * (v - 4))) + kc * 32;
                a[2 * v]     = (_Float16)mlpin[(size_t)nodeA * HID + k];
                a[2 * v + 1] = (_Float16)mlpin[(size_t)nodeA * HID + k + 1];
                const int kb = k;
                b[2 * v]     = (_Float16)W1[(size_t)kb * 128 + c0 + mn];
                b[2 * v + 1] = (_Float16)W1[(size_t)(kb + 1) * 128 + c0 + mn];
            }
            acc = __builtin_amdgcn_wmma_f32_16x16x32_f16(false, a, false, b,
                                                         (short)0, acc, false, false);
        }
        #pragma unroll
        for (int v = 0; v < 8; ++v) {
            const int row = v + grp * 8;
            zf[row][c0 + mn] = acc[v] + b1[c0 + mn];
        }
    }
    __syncthreads();

    // ---------------- LayerNorm(128) + ReLU -> f16, wave-parallel ----------------
    // Wave w owns rows 2w and 2w+1; each lane covers 4 columns; cross-lane
    // reduction via __shfl_xor (wave32).
    #pragma unroll
    for (int rr = 0; rr < 2; ++rr) {
        const int row = wave * 2 + rr;
        float s = 0.0f;
        #pragma unroll
        for (int q = 0; q < 4; ++q) s += zf[row][lane * 4 + q];
        #pragma unroll
        for (int off = 16; off > 0; off >>= 1) s += __shfl_xor(s, off, 32);
        const float mu = s * (1.0f / 128.0f);

        float vs = 0.0f;
        #pragma unroll
        for (int q = 0; q < 4; ++q) {
            const float d = zf[row][lane * 4 + q] - mu;
            vs += d * d;
        }
        #pragma unroll
        for (int off = 16; off > 0; off >>= 1) vs += __shfl_xor(vs, off, 32);
        const float rs = rsqrtf(vs * (1.0f / 128.0f) + LN_EPS);

        #pragma unroll
        for (int q = 0; q < 4; ++q) {
            const int c = lane * 4 + q;
            const float val = (zf[row][c] - mu) * rs * lng[c] + lnb[c];
            zh[row][c] = (_Float16)fmaxf(val, 0.0f);
        }
    }
    __syncthreads();

    // ---------------- GEMM2: zln[16x128] @ W2[128x64] + residuals ----------------
    if (wave < 4) {
        const int c0 = wave * 16;
        v8f acc = {};
        #pragma unroll
        for (int kc = 0; kc < 4; ++kc) {
            v16h a, b;
            #pragma unroll
            for (int v = 0; v < 8; ++v) {
                const int k = ((v < 4) ? (grp * 8 + 2 * v)
                                       : (16 + grp * 8 + 2 * (v - 4))) + kc * 32;
                a[2 * v]     = zh[mn][k];
                a[2 * v + 1] = zh[mn][k + 1];
                const int kb = k;
                b[2 * v]     = (_Float16)W2[(size_t)kb * HID + c0 + mn];
                b[2 * v + 1] = (_Float16)W2[(size_t)(kb + 1) * HID + c0 + mn];
            }
            acc = __builtin_amdgcn_wmma_f32_16x16x32_f16(false, a, false, b,
                                                         (short)0, acc, false, false);
        }
        #pragma unroll
        for (int v = 0; v < 8; ++v) {
            const int row  = v + grp * 8;
            const int node = blockIdx.x * 16 + row;
            const int col  = c0 + mn;
            float val = acc[v] + b2[col];
            if (r1) val += r1[(size_t)node * HID + col];
            if (r2) val += r2[(size_t)node * HID + col];
            out[(size_t)node * HID + col] = fmaxf(val, 0.0f);
        }
    }
}

// ---------------------------------------------------------------------------
// Input order (setup_inputs dict order, depth-first):
//  0 x  1 edge_index  2 edge_attr  3 face_grid  4 edge_grid
//  5..12 enc_params: face{W,b} face_grid{W,b} edge{W,b} edge_grid{W,b}
//  13+7L..: conv_params[L]: mlp1.W mlp1.b ln_g ln_b mlp2.W mlp2.b t
// ---------------------------------------------------------------------------
extern "C" void kernel_launch(void* const* d_in, const int* in_sizes, int n_in,
                              void* d_out, int out_size, void* d_ws, size_t ws_size,
                              hipStream_t stream)
{
    (void)in_sizes; (void)n_in; (void)out_size; (void)ws_size;

    const float* x    = (const float*)d_in[0];
    const int*   ei   = (const int*)  d_in[1];
    const float* ea   = (const float*)d_in[2];
    const float* fg   = (const float*)d_in[3];
    const float* eg   = (const float*)d_in[4];
    const float* WfW  = (const float*)d_in[5];
    const float* Wfb  = (const float*)d_in[6];
    const float* WgW  = (const float*)d_in[7];
    const float* Wgb  = (const float*)d_in[8];
    const float* WeW  = (const float*)d_in[9];
    const float* Web  = (const float*)d_in[10];
    const float* WegW = (const float*)d_in[11];
    const float* Wegb = (const float*)d_in[12];

    float* ws   = (float*)d_ws;
    float* e    = ws;                                  // E*64
    float* h0   = e    + (size_t)N_EDGES * HID;        // N*64 each below
    float* x1   = h0   + (size_t)N_NODES * HID;
    float* x2   = x1   + (size_t)N_NODES * HID;
    float* maxb = x2   + (size_t)N_NODES * HID;        // reused as mlp_in after edge passes
    float* numb = maxb + (size_t)N_NODES * HID;
    float* denb = numb + (size_t)N_NODES * HID;

    enc_kernel<<<N_NODES, 64, 0, stream>>>(x,  fg, WfW, Wfb, WgW,  Wgb,  h0);
    enc_kernel<<<N_EDGES, 64, 0, stream>>>(ea, eg, WeW, Web, WegW, Wegb, e);

    const float* layer_in[3]  = { h0, x1, x2 };
    float*       layer_out[3] = { x1, x2, (float*)d_out };
    const float* res1[3]      = { nullptr, x1, x2 };
    const float* res2[3]      = { nullptr, nullptr, x1 };

    const int NF = N_NODES * HID;
    for (int L = 0; L < 3; ++L) {
        const float* W1  = (const float*)d_in[13 + 7 * L + 0];
        const float* b1  = (const float*)d_in[13 + 7 * L + 1];
        const float* lng = (const float*)d_in[13 + 7 * L + 2];
        const float* lnb = (const float*)d_in[13 + 7 * L + 3];
        const float* W2  = (const float*)d_in[13 + 7 * L + 4];
        const float* b2  = (const float*)d_in[13 + 7 * L + 5];
        const float* t   = (const float*)d_in[13 + 7 * L + 6];

        clear_kernel<<<(NF + 255) / 256, 256, 0, stream>>>(maxb, numb, denb, NF);
        edge_max_kernel<<<N_EDGES, 64, 0, stream>>>(layer_in[L], e, ei, t, maxb);
        edge_sum_kernel<<<N_EDGES, 64, 0, stream>>>(layer_in[L], e, ei, t, maxb, numb, denb);
        // maxb is dead now -> reuse as the finalized MLP input buffer.
        aggr_kernel<<<(NF + 255) / 256, 256, 0, stream>>>(numb, denb, layer_in[L], maxb, NF);
        conv_mlp_kernel<<<N_NODES / 16, 256, 0, stream>>>(maxb,
                                                          W1, b1, lng, lnb, W2, b2,
                                                          res1[L], res2[L], layer_out[L]);
    }
}
